// SelfAttention_13975823581888
// MI455X (gfx1250) — matrix operand order repaired and checked by reading, not compiled, not run on hardware
//
#include <hip/hip_runtime.h>
#include <hip/hip_bf16.h>
#include <stdint.h>

#define EMBED 512
#define HEADS 8
#define HDIM  64
#define NB    2
#define SEQ   3072

typedef __attribute__((ext_vector_type(16))) _Float16 v16h;
typedef __attribute__((ext_vector_type(8)))  float    v8f;
typedef __attribute__((ext_vector_type(4)))  uint32_t u32x4;
typedef __attribute__((ext_vector_type(4)))  int      i32x4;

union AF { v16h v; uint32_t u[8]; _Float16 h[16]; };
union BF { v16h v; u32x4 q4[2]; _Float16 h[16]; };
union H8 { u32x4 q4; _Float16 h[8]; };

// ---- CDNA5 async-to-LDS path (verified: builtin exists & selects) ----
#if __has_builtin(__builtin_amdgcn_global_load_async_to_lds_b128) && \
    __has_builtin(__builtin_amdgcn_s_wait_asynccnt)
#define HAVE_ASYNC 1
typedef __attribute__((address_space(1))) i32x4* gv4p;  // global V4i*
typedef __attribute__((address_space(3))) i32x4* lv4p;  // LDS V4i*
__device__ __forceinline__ void async_b128(const void* g, void* l) {
    // lds operand: low 32 bits of the flat shared address == LDS byte offset
    __builtin_amdgcn_global_load_async_to_lds_b128(
        (gv4p)(uintptr_t)g, (lv4p)(uintptr_t)(uint32_t)(uintptr_t)l, 0, 0);
}
#else
#define HAVE_ASYNC 0
#endif

__device__ __forceinline__ void wait_ds0() {
#if __has_builtin(__builtin_amdgcn_s_wait_dscnt)
    __builtin_amdgcn_s_wait_dscnt(0);
#else
    __syncthreads();
#endif
}

// ds_swizzle_b32 group-of-32 xor butterfly: imm = (xor<<10) | and_mask(0x1f).
// Immediate-pattern lane exchange: no per-lane address math (vs __shfl_xor).
template <int XORM>
__device__ __forceinline__ float swz_xor(float x) {
    int r = __builtin_amdgcn_ds_swizzle(__float_as_int(x), (XORM << 10) | 0x1f);
    return __int_as_float(r);
}
__device__ __forceinline__ float red_max16(float x) {
    x = fmaxf(x, swz_xor<1>(x));
    x = fmaxf(x, swz_xor<2>(x));
    x = fmaxf(x, swz_xor<4>(x));
    x = fmaxf(x, swz_xor<8>(x));
    return x;
}
__device__ __forceinline__ float red_sum16(float x) {
    x += swz_xor<1>(x);
    x += swz_xor<2>(x);
    x += swz_xor<4>(x);
    x += swz_xor<8>(x);
    return x;
}

// A-matrix 16x32 layout (contiguous k, matching B fills): pair p holds K = {2p,2p+1};
// lanes 16-31 add +16.
__device__ __forceinline__ int a_pair_k(int p, int lane) {
    return 2 * p + ((lane & 16) ? 16 : 0);
}

static __device__ __forceinline__ v8f wmma_f16(v16h a, v16h b, v8f c) {
    // D = A(16x32 f16) x B(32x16 f16) + C(16x16 f32)
    return __builtin_amdgcn_wmma_f32_16x16x32_f16(false, a, false, b, (short)0, c,
                                                  false, false);
}

// ---------------------------------------------------------------- Wfc -> f16
__global__ void wfc_to_f16(const float* __restrict__ w, _Float16* __restrict__ o) {
    int i = blockIdx.x * blockDim.x + threadIdx.x;
    for (; i < EMBED * EMBED; i += gridDim.x * blockDim.x) o[i] = (_Float16)w[i];
}

// ------------------------------------------------- per-head projections (f32)
// v = v_in @ Wv^T ; k = v @ Wv^T (faithful bug: `keys` input ignored);
// q = (q_in @ Wq^T) * 1/sqrt(512).
// Outputs: q16,k16 head-major [n][h][s][64]; V emitted pre-transposed
// vT16 [n][h][d][s] so attention can stage it with contiguous b128 copies.
__global__ __launch_bounds__(256) void proj_kernel(
    const float* __restrict__ vin, const float* __restrict__ qin,
    const float* __restrict__ Wv, const float* __restrict__ Wq,
    _Float16* __restrict__ q16, _Float16* __restrict__ k16,
    _Float16* __restrict__ vT16) {
    __shared__ float wv[64 * 64];
    __shared__ float wq[64 * 64];
    __shared__ float vt[64 * 68];

    const int tid = threadIdx.x;
    const int SB = SEQ / 64;
    const int nh = blockIdx.x / SB, sblk = blockIdx.x % SB;
    const int n = nh / HEADS, h = nh % HEADS;

    for (int i = tid; i < 64 * 64; i += 256) { wv[i] = Wv[i]; wq[i] = Wq[i]; }
    __syncthreads();

    const int tok  = tid >> 2;            // 0..63
    const int eseg = (tid & 3) << 4;      // 0,16,32,48
    const int s = sblk * 64 + tok;
    const size_t grow = ((size_t)n * SEQ + s) * EMBED + h * HDIM;
    const size_t hrow = ((size_t)(n * HEADS + h) * SEQ + s) * HDIM;
    const float QS = 0.04419417382415922f;  // 1/sqrt(512)

    float x[64];
    // ---- v projection (kept in LDS as f32 for the k = Wv(v) chain) ----
    {
        const float4* src = (const float4*)(vin + grow);
#pragma unroll
        for (int i = 0; i < 16; ++i) {
            float4 t = src[i];
            x[4*i] = t.x; x[4*i+1] = t.y; x[4*i+2] = t.z; x[4*i+3] = t.w;
        }
    }
#pragma unroll
    for (int eo = 0; eo < 16; ++eo) {
        const float4* wr = (const float4*)&wv[(eseg + eo) * 64];
        float acc = 0.f;
#pragma unroll
        for (int d4 = 0; d4 < 16; ++d4) {
            float4 w = wr[d4];
            acc += x[4*d4]*w.x + x[4*d4+1]*w.y + x[4*d4+2]*w.z + x[4*d4+3]*w.w;
        }
        vt[tok * 68 + eseg + eo] = acc;
    }
    __syncthreads();
    // ---- k = v @ Wv^T ----
    {
        const float4* src = (const float4*)&vt[tok * 68];
#pragma unroll
        for (int i = 0; i < 16; ++i) {
            float4 t = src[i];
            x[4*i] = t.x; x[4*i+1] = t.y; x[4*i+2] = t.z; x[4*i+3] = t.w;
        }
    }
#pragma unroll
    for (int eo = 0; eo < 16; ++eo) {
        const float4* wr = (const float4*)&wv[(eseg + eo) * 64];
        float acc = 0.f;
#pragma unroll
        for (int d4 = 0; d4 < 16; ++d4) {
            float4 w = wr[d4];
            acc += x[4*d4]*w.x + x[4*d4+1]*w.y + x[4*d4+2]*w.z + x[4*d4+3]*w.w;
        }
        k16[hrow + eseg + eo] = (_Float16)acc;
    }
    // ---- V transposed emit: vT16[(nh*64 + d)*SEQ + s], coalesced b128 ----
    {
        const int d = tid >> 2, sc = tid & 3;
        H8 o0, o1;
#pragma unroll
        for (int i = 0; i < 8; ++i) o0.h[i] = (_Float16)vt[(sc * 16 + i) * 68 + d];
#pragma unroll
        for (int i = 0; i < 8; ++i) o1.h[i] = (_Float16)vt[(sc * 16 + 8 + i) * 68 + d];
        _Float16* dst = vT16 + ((size_t)(n * HEADS + h) * HDIM + d) * SEQ +
                        sblk * 64 + sc * 16;
        *(u32x4*)dst = o0.q4;
        *(u32x4*)(dst + 8) = o1.q4;
    }
    // ---- q projection (1/sqrt(512) folded in) ----
    {
        const float4* src = (const float4*)(qin + grow);
#pragma unroll
        for (int i = 0; i < 16; ++i) {
            float4 t = src[i];
            x[4*i] = t.x; x[4*i+1] = t.y; x[4*i+2] = t.z; x[4*i+3] = t.w;
        }
    }
#pragma unroll
    for (int eo = 0; eo < 16; ++eo) {
        const float4* wr = (const float4*)&wq[(eseg + eo) * 64];
        float acc = 0.f;
#pragma unroll
        for (int d4 = 0; d4 < 16; ++d4) {
            float4 w = wr[d4];
            acc += x[4*d4]*w.x + x[4*d4+1]*w.y + x[4*d4+2]*w.z + x[4*d4+3]*w.w;
        }
        q16[hrow + eseg + eo] = (_Float16)(acc * QS);
    }
}

// -------------------------------------------------------- flash attention
// 8 waves / block; wave owns 16 query rows; block = 128 queries for one (n,h).
// Double-buffered K/V^T tiles staged with async-to-LDS (ASYNCcnt).
__global__ __launch_bounds__(256) void attn_kernel(
    const _Float16* __restrict__ q16, const _Float16* __restrict__ k16,
    const _Float16* __restrict__ vT16, const unsigned char* __restrict__ mask,
    _Float16* __restrict__ att) {
    __shared__ _Float16 Kt[2][64 * 72];    // [key][d], padded
    __shared__ _Float16 Vt[2][64 * 72];    // [d][l]  (pre-transposed), padded
    __shared__ _Float16 Ps[8 * 16 * 72];   // per-wave P scratch (C->A relayout)

    const int tid = threadIdx.x;
    const int lane = tid & 31;
    const int wave = tid >> 5;
    const int QB = SEQ / 128;
    const int nh = blockIdx.x / QB, qblk = blockIdx.x % QB;
    const int n = nh / HEADS, h = nh % HEADS;
    const int ln  = lane & 15;
    const int hi  = lane & 16;   // 0 or 16
    const int hi8 = hi >> 1;     // 0 or 8

    const _Float16* Qh = q16 + (size_t)(n * HEADS + h) * SEQ * HDIM;
    const _Float16* Kh = k16 + (size_t)(n * HEADS + h) * SEQ * HDIM;
    const _Float16* VTh = vT16 + (size_t)(n * HEADS + h) * HDIM * SEQ;
    const unsigned char* mrow = mask + (size_t)n * SEQ;
    const int qrow0 = qblk * 128 + wave * 16;

    // staging mapping: thread -> row r (key for K, dim d for V^T), 16-col chunk
    const int sr = tid >> 2, sc0 = (tid & 3) << 4;
    const _Float16* kcur = Kh + (size_t)sr * HDIM + sc0;  // advances 64*HDIM/iter
    const _Float16* vcur = VTh + (size_t)sr * SEQ + sc0;  // advances 64/iter
    _Float16* kdst[2] = { &Kt[0][sr * 72 + sc0], &Kt[1][sr * 72 + sc0] };
    _Float16* vdst[2] = { &Vt[0][sr * 72 + sc0], &Vt[1][sr * 72 + sc0] };

    // Q A-fragments (16 rows x 64 dims), loaded once
    AF aq[2];
#pragma unroll
    for (int f = 0; f < 2; ++f) {
        const _Float16* qr = Qh + (size_t)(qrow0 + ln) * HDIM + 32 * f;
#pragma unroll
        for (int p = 0; p < 8; ++p)
            aq[f].u[p] = *(const uint32_t*)(qr + a_pair_k(p, lane));
    }

    const v8f vz = {0.f, 0.f, 0.f, 0.f, 0.f, 0.f, 0.f, 0.f};
    v8f o[4];
#pragma unroll
    for (int j = 0; j < 4; ++j) o[j] = vz;
    float m_i[8], l_i[8];
#pragma unroll
    for (int i = 0; i < 8; ++i) { m_i[i] = -1e30f; l_i[i] = 0.f; }

#if HAVE_ASYNC
    // prologue: stage tile 0 into buffer 0
    async_b128(kcur, kdst[0]);
    async_b128(kcur + 8, kdst[0] + 8);
    async_b128(vcur, vdst[0]);
    async_b128(vcur + 8, vdst[0] + 8);
#endif

    for (int kb = 0; kb < SEQ; kb += 64) {
        const int buf = (kb >> 6) & 1;
        if (kb) __syncthreads();  // all waves done with buffer we're about to fill
#if HAVE_ASYNC
        if (kb + 64 < SEQ) {      // issue next tile into the other buffer
            const _Float16* kn = kcur + 64 * HDIM;
            const _Float16* vn = vcur + 64;
            async_b128(kn, kdst[buf ^ 1]);
            async_b128(kn + 8, kdst[buf ^ 1] + 8);
            async_b128(vn, vdst[buf ^ 1]);
            async_b128(vn + 8, vdst[buf ^ 1] + 8);
            __builtin_amdgcn_s_wait_asynccnt(4);  // current tile's 4 copies done
        } else {
            __builtin_amdgcn_s_wait_asynccnt(0);
        }
#else
        {
            *(u32x4*)kdst[buf] = *(const u32x4*)kcur;
            *(u32x4*)(kdst[buf] + 8) = *(const u32x4*)(kcur + 8);
            *(u32x4*)vdst[buf] = *(const u32x4*)vcur;
            *(u32x4*)(vdst[buf] + 8) = *(const u32x4*)(vcur + 8);
            if (kb + 64 < SEQ) {
                __builtin_prefetch(kcur + 64 * HDIM, 0, 1);
                __builtin_prefetch(vcur + 64, 0, 1);
            }
        }
#endif
        kcur += 64 * HDIM;
        vcur += 64;
        __syncthreads();  // staged tile visible to all waves

        // ---- S = Q K^T over 4 key sub-tiles of 16 ----
        v8f sc[4];
#pragma unroll
        for (int j = 0; j < 4; ++j) {
            BF b0, b1;
            const _Float16* kr = &Kt[buf][(j * 16 + ln) * 72 + hi];
            b0.q4[0] = *(const u32x4*)kr;        b0.q4[1] = *(const u32x4*)(kr + 8);
            b1.q4[0] = *(const u32x4*)(kr + 32); b1.q4[1] = *(const u32x4*)(kr + 40);
            v8f acc = vz;
            acc = wmma_f16(aq[0].v, b0.v, acc);
            acc = wmma_f16(aq[1].v, b1.v, acc);
            if (mrow[kb + j * 16 + ln] == 0) {
#pragma unroll
                for (int i = 0; i < 8; ++i) acc[i] = -4.4194174e8f;  // -1e10/sqrt(512)
            }
            sc[j] = acc;
        }

        // ---- online softmax update (ds_swizzle xor-16 butterflies) ----
#pragma unroll
        for (int i = 0; i < 8; ++i) {
            float pm = fmaxf(fmaxf(sc[0][i], sc[1][i]), fmaxf(sc[2][i], sc[3][i]));
            pm = red_max16(pm);
            const float mn = fmaxf(m_i[i], pm);
            const float alpha = __expf(m_i[i] - mn);
            m_i[i] = mn;
            float r = 0.f;
#pragma unroll
            for (int j = 0; j < 4; ++j) {
                float e = __expf(sc[j][i] - mn);
                sc[j][i] = e;
                r += e;
            }
            r = red_sum16(r);
            l_i[i] = l_i[i] * alpha + r;
            o[0][i] *= alpha; o[1][i] *= alpha; o[2][i] *= alpha; o[3][i] *= alpha;
        }

        // ---- P: C-layout -> wave-private LDS -> A-layout ----
        _Float16* Pw = &Ps[wave * 16 * 72];
#pragma unroll
        for (int j = 0; j < 4; ++j)
#pragma unroll
            for (int i = 0; i < 8; ++i)
                Pw[(i + hi8) * 72 + j * 16 + ln] = (_Float16)sc[j][i];
        wait_ds0();  // wave-private scratch: DS-counter fence, no block barrier
        AF ap[2];
#pragma unroll
        for (int f = 0; f < 2; ++f) {
            const _Float16* pr = Pw + (size_t)ln * 72 + 32 * f;
#pragma unroll
            for (int p = 0; p < 8; ++p)
                ap[f].u[p] = *(const uint32_t*)(pr + a_pair_k(p, lane));
        }

        // ---- O += P V ----
#pragma unroll
        for (int j = 0; j < 4; ++j) {
            BF b0, b1;
            const _Float16* vr = &Vt[buf][(j * 16 + ln) * 72 + hi];
            b0.q4[0] = *(const u32x4*)vr;        b0.q4[1] = *(const u32x4*)(vr + 8);
            b1.q4[0] = *(const u32x4*)(vr + 32); b1.q4[1] = *(const u32x4*)(vr + 40);
            o[j] = wmma_f16(ap[0].v, b0.v, o[j]);
            o[j] = wmma_f16(ap[1].v, b1.v, o[j]);
        }
    }

    // ---- normalize & emit f16 [n][s][h*64+d] ----
#pragma unroll
    for (int i = 0; i < 8; ++i) {
        const float inv = l_i[i] > 0.f ? 1.f / l_i[i] : 0.f;
        const int q = qrow0 + i + hi8;
        _Float16* dst = att + ((size_t)n * SEQ + q) * EMBED + h * HDIM;
#pragma unroll
        for (int j = 0; j < 4; ++j) dst[j * 16 + ln] = (_Float16)(o[j][i] * inv);
    }
}

// -------------------------------------------------- FC: out = att @ Wfc^T + b
__global__ __launch_bounds__(256) void fc_kernel(
    const _Float16* __restrict__ att, const _Float16* __restrict__ wfc,
    const float* __restrict__ bfc, float* __restrict__ out) {
    const int tid = threadIdx.x, lane = tid & 31, wave = tid >> 5;
    const int ln = lane & 15, hi = lane & 16, hi8 = hi >> 1;
    const int row0 = blockIdx.x * 32 + (wave >> 2) * 16;
    const int e0 = (wave & 3) * 128;

    const v8f vz = {0.f, 0.f, 0.f, 0.f, 0.f, 0.f, 0.f, 0.f};
    v8f o[8];
#pragma unroll
    for (int t = 0; t < 8; ++t) o[t] = vz;

    for (int c0 = 0; c0 < EMBED; c0 += 32) {
        AF a;
        const _Float16* ar = att + (size_t)(row0 + ln) * EMBED + c0;
#pragma unroll
        for (int p = 0; p < 8; ++p)
            a.u[p] = *(const uint32_t*)(ar + a_pair_k(p, lane));
#pragma unroll
        for (int t = 0; t < 8; ++t) {
            BF b;  // B[c][e] = Wfc[e][c]; lane -> column e, contiguous c run
            const _Float16* wr = wfc + (size_t)(e0 + t * 16 + ln) * EMBED + c0 + hi;
            b.q4[0] = *(const u32x4*)wr;
            b.q4[1] = *(const u32x4*)(wr + 8);
            o[t] = wmma_f16(a.v, b.v, o[t]);
        }
    }
#pragma unroll
    for (int t = 0; t < 8; ++t) {
        const int e = e0 + t * 16 + ln;
        const float bias = bfc[e];
#pragma unroll
        for (int i = 0; i < 8; ++i)
            out[(size_t)(row0 + i + hi8) * EMBED + e] = o[t][i] + bias;
    }
}

extern "C" void kernel_launch(void* const* d_in, const int* in_sizes, int n_in,
                              void* d_out, int out_size, void* d_ws, size_t ws_size,
                              hipStream_t stream) {
    (void)in_sizes; (void)n_in; (void)out_size; (void)ws_size;
    const float* values = (const float*)d_in[0];
    // d_in[1] (`keys`) intentionally ignored — faithful to reference bug.
    const float* query = (const float*)d_in[2];
    const unsigned char* mask = (const unsigned char*)d_in[3];
    const float* Wv = (const float*)d_in[4];
    const float* Wq = (const float*)d_in[5];
    const float* Wfc = (const float*)d_in[6];
    const float* bfc = (const float*)d_in[7];
    float* out = (float*)d_out;

    const size_t NHSD = (size_t)NB * HEADS * SEQ * HDIM;  // == N*S*EMBED
    _Float16* q16 = (_Float16*)d_ws;
    _Float16* k16 = q16 + NHSD;
    _Float16* vT16 = k16 + NHSD;
    _Float16* att = vT16 + NHSD;
    _Float16* wfc16 = att + NHSD;

    wfc_to_f16<<<256, 256, 0, stream>>>(Wfc, wfc16);
    proj_kernel<<<NB * HEADS * (SEQ / 64), 256, 0, stream>>>(values, query, Wv, Wq,
                                                             q16, k16, vT16);
    attn_kernel<<<NB * HEADS * (SEQ / 128), 256, 0, stream>>>(q16, k16, vT16, mask,
                                                              att);
    fc_kernel<<<(NB * SEQ) / 32, 256, 0, stream>>>(att, wfc16, bfc, out);
}
